// LSSTransform_Lite_609885356386
// MI455X (gfx1250) — matrix operand order repaired and checked by reading, not compiled, not run on hardware
//
#include <hip/hip_runtime.h>
#include <cstddef>

typedef __attribute__((ext_vector_type(16))) _Float16 v16h;
typedef __attribute__((ext_vector_type(8)))  float    v8f;

#define D_BINS 118
#define C_OUT  80
#define O_TOT  198          // D_BINS + C_OUT
#define O_PAD  208          // padded to 13*16 so WMMA stores are branchless
#define CIN    256
#define FH     32
#define FW     88
#define HW     (FH*FW)      // 2816
#define NCAM   6
#define NXV    360
#define NYV    360
#define PLANE  (NXV*NYV)    // 129600
#define O_TILES  13         // 208/16
#define HWG      22         // groups of 128 hw-cols (4 waves x 2 tiles x 16)

// ---------------------------------------------------------------------------
// Kernel 0: per-camera transform precompute + bias padding.
// camp layout per cam (24 floats):
//   [0..8]  combine = rots @ inv(intrins)   (row major)
//   [9..17] inv(post_rots)                  (row major)
//   [18..20] trans
//   [21..23] post_trans
// bpad: 208 floats, bias zero-padded past 198.
// ---------------------------------------------------------------------------
__device__ inline void inv3(const float* a, float* o) {
    float a00=a[0],a01=a[1],a02=a[2],a10=a[3],a11=a[4],a12=a[5],a20=a[6],a21=a[7],a22=a[8];
    float det = a00*(a11*a22-a12*a21) - a01*(a10*a22-a12*a20) + a02*(a10*a21-a11*a20);
    float id = 1.0f/det;
    o[0]=(a11*a22-a12*a21)*id; o[1]=(a02*a21-a01*a22)*id; o[2]=(a01*a12-a02*a11)*id;
    o[3]=(a12*a20-a10*a22)*id; o[4]=(a00*a22-a02*a20)*id; o[5]=(a02*a10-a00*a12)*id;
    o[6]=(a10*a21-a11*a20)*id; o[7]=(a01*a20-a00*a21)*id; o[8]=(a00*a11-a01*a10)*id;
}

__global__ __launch_bounds__(256)
void prep_kernel(const float* __restrict__ rots,
                 const float* __restrict__ trans,
                 const float* __restrict__ intrins,
                 const float* __restrict__ prots,
                 const float* __restrict__ ptrans,
                 const float* __restrict__ bias,
                 float* __restrict__ camp,
                 float* __restrict__ bpad) {
    const int t = threadIdx.x;
    if (t < O_PAD)
        bpad[t] = (t < O_TOT) ? bias[t] : 0.0f;
    if (t < NCAM) {
        float invK[9], comb[9], invPR[9];
        inv3(intrins + 9*t, invK);
        inv3(prots   + 9*t, invPR);
        const float* R = rots + 9*t;
        for (int i = 0; i < 3; ++i)
            for (int j = 0; j < 3; ++j)
                comb[3*i+j] = R[3*i+0]*invK[0*3+j] + R[3*i+1]*invK[1*3+j] + R[3*i+2]*invK[2*3+j];
        float* cp = camp + 24*t;
        for (int i = 0; i < 9; ++i) { cp[i] = comb[i]; cp[9+i] = invPR[i]; }
        for (int i = 0; i < 3; ++i) { cp[18+i] = trans[3*t+i]; cp[21+i] = ptrans[3*t+i]; }
    }
}

// ---------------------------------------------------------------------------
// Kernel 1: WMMA GEMM  x[cam,o,hw] = sum_c W[o,c]*img[cam,c,hw] + b[o]
// One wave owns a 16(M) x 32(N) output strip: one A fragment shared by two
// B fragments -> 2 x v_wmma_f32_16x16x32_f16 per K-chunk, K=256 in 8 chunks.
// Output O dimension padded to 208 so all stores are branchless (constant
// immediate offsets r*HW*4).
// ---------------------------------------------------------------------------
__global__ __launch_bounds__(128)
void gemm_wmma_kernel(const float* __restrict__ img,
                      const float* __restrict__ W,
                      const float* __restrict__ bpad,
                      float* __restrict__ xws) {
    const int wave = threadIdx.x >> 5;
    const int lane = threadIdx.x & 31;
    const int bid  = blockIdx.x;
    const int cam  = bid / (O_TILES * HWG);
    const int rem  = bid % (O_TILES * HWG);
    const int ot   = rem / HWG;
    const int hwg  = rem % HWG;
    const int hw0  = hwg * 128 + wave * 32;      // 32 columns per wave

    const float* imgc = img + (size_t)cam * CIN * HW;
    const int col0 = hw0 + (lane & 15);
    const int col1 = col0 + 16;
    const int rowq = ot * 16 + (lane & 15);
    const int arow = (rowq < O_TOT) ? rowq : (O_TOT - 1);   // clamp, keep EXEC full
    const float* Wrow = W + (size_t)arow * CIN;

    // k order this code uses for v16h fragments (16x32 A, 32x16 B), same for A and B:
    //  A: lanes 0-15 row M=lane,     halfs j   -> K=j
    //     lanes 16-31 row M=lane-16, halfs j   -> K=j+16
    //  B: lanes 0-15 col N=lane,     halfs j   -> K=j
    //     lanes 16-31 col N=lane-16, halfs j   -> K=j+16
    const int kA_off = (lane < 16) ? 0 : 16;
    const int kB_off = (lane < 16) ? 0 : 16;

    v8f acc0 = {};
    v8f acc1 = {};
    for (int kk = 0; kk < CIN; kk += 32) {
        v16h a, b0, b1;
#pragma unroll
        for (int j = 0; j < 16; ++j) {
            const int ka = kk + j + kA_off;
            a[j] = (_Float16)Wrow[ka];
            const float* brow = imgc + (size_t)(kk + j + kB_off) * HW;
            b0[j] = (_Float16)brow[col0];
            b1[j] = (_Float16)brow[col1];
        }
        acc0 = __builtin_amdgcn_wmma_f32_16x16x32_f16(
                   false, a, false, b0, (short)0, acc0, false, false);
        acc1 = __builtin_amdgcn_wmma_f32_16x16x32_f16(
                   false, a, false, b1, (short)0, acc1, false, false);
    }

    // C/D layout: VGPR r -> M = r + 8*(lane>=16), N = lane&15
    const int obase = ot * 16 + ((lane >> 4) << 3);
    float* p0 = xws + ((size_t)cam * O_PAD + obase) * HW + col0;
    const float* bb = bpad + obase;
#pragma unroll
    for (int r = 0; r < 8; ++r) {
        const float bv = bb[r];
        p0[(size_t)r * HW]      = acc0[r] + bv;
        p0[(size_t)r * HW + 16] = acc1[r] + bv;
    }
}

// ---------------------------------------------------------------------------
// Kernel 2: fused softmax(depth) + geometry + outer-product scatter-add.
// One 128-thread block per pixel (cam,h,w). vol is never materialized:
// 160M hardware fp32 atomics into the 41.5MB (L2-resident) BEV grid.
// ---------------------------------------------------------------------------
__global__ __launch_bounds__(128)
void splat_kernel(const float* __restrict__ xws,
                  const float* __restrict__ camp,
                  float* __restrict__ out) {
    __shared__ float sdep[D_BINS];
    __shared__ float sfeat[C_OUT];
    __shared__ int   svox[D_BINS];
    __shared__ float sred[128];

    const int t   = threadIdx.x;
    const int pix = blockIdx.x;          // cam*HW + hw
    const int cam = pix / HW;
    const int hw  = pix % HW;
    const int h   = hw / FW;
    const int w   = hw % FW;

    const float* xc = xws + (size_t)cam * O_PAD * HW + hw;

    // --- softmax over 118 depth logits ---
    float m = -3.4e38f;
    for (int d = t; d < D_BINS; d += 128) {
        float v = xc[(size_t)d * HW];
        sdep[d] = v;
        m = fmaxf(m, v);
    }
    sred[t] = m;
    __syncthreads();
    for (int s = 64; s > 0; s >>= 1) {
        if (t < s) sred[t] = fmaxf(sred[t], sred[t + s]);
        __syncthreads();
    }
    m = sred[0];
    __syncthreads();
    float sum = 0.f;
    for (int d = t; d < D_BINS; d += 128) {
        float e = __expf(sdep[d] - m);
        sdep[d] = e;
        sum += e;
    }
    sred[t] = sum;
    __syncthreads();
    for (int s = 64; s > 0; s >>= 1) {
        if (t < s) sred[t] += sred[t + s];
        __syncthreads();
    }
    const float rinv = 1.0f / sred[0];
    __syncthreads();

    // --- context features ---
    for (int c = t; c < C_OUT; c += 128)
        sfeat[c] = xc[(size_t)(D_BINS + c) * HW];

    // --- frustum -> ego geometry, one depth bin per thread ---
    const float* cp = camp + 24 * cam;
    if (t < D_BINS) {
        const float fx = (float)w * (703.0f / 87.0f);
        const float fy = (float)h * (255.0f / 31.0f);
        const float dv = 1.0f + 0.5f * (float)t;
        const float u  = fx - cp[21];
        const float v2 = fy - cp[22];
        const float z0 = dv - cp[23];
        const float p1x = cp[ 9]*u + cp[10]*v2 + cp[11]*z0;
        const float p1y = cp[12]*u + cp[13]*v2 + cp[14]*z0;
        const float p1z = cp[15]*u + cp[16]*v2 + cp[17]*z0;
        const float q0 = p1x * p1z, q1 = p1y * p1z, q2 = p1z;
        const float gx = cp[0]*q0 + cp[1]*q1 + cp[2]*q2 + cp[18];
        const float gy = cp[3]*q0 + cp[4]*q1 + cp[5]*q2 + cp[19];
        const float gz = cp[6]*q0 + cp[7]*q1 + cp[8]*q2 + cp[20];
        const int cx = (int)floorf((gx + 54.0f) / 0.3f);
        const int cy = (int)floorf((gy + 54.0f) / 0.3f);
        const int cz = (int)floorf((gz + 10.0f) / 20.0f);
        const bool kept = (cx >= 0) && (cx < NXV) && (cy >= 0) && (cy < NYV) && (cz == 0);
        svox[t] = kept ? (cx * NYV + cy) : -1;
    }
    __syncthreads();

    // --- scatter:  out[c, cx, cy] += depth[d] * feat[c] ---
    for (int idx = t; idx < D_BINS * C_OUT; idx += 128) {
        const int d  = idx / C_OUT;
        const int c  = idx - d * C_OUT;
        const int vb = svox[d];
        if (vb >= 0)
            unsafeAtomicAdd(out + (size_t)c * PLANE + vb, sdep[d] * rinv * sfeat[c]);
    }
}

// ---------------------------------------------------------------------------
extern "C" void kernel_launch(void* const* d_in, const int* in_sizes, int n_in,
                              void* d_out, int out_size, void* d_ws, size_t ws_size,
                              hipStream_t stream) {
    const float* img     = (const float*)d_in[0];
    const float* rots    = (const float*)d_in[1];
    const float* trans   = (const float*)d_in[2];
    const float* intrins = (const float*)d_in[3];
    const float* prots   = (const float*)d_in[4];
    const float* ptrans  = (const float*)d_in[5];
    const float* W       = (const float*)d_in[6];
    const float* bias    = (const float*)d_in[7];
    float* out = (float*)d_out;

    float* xws  = (float*)d_ws;                                   // 6*208*2816 f32 (~14 MB)
    float* camp = xws + (size_t)NCAM * O_PAD * HW;                // 6*24 f32
    float* bpad = camp + NCAM * 24;                               // 208 f32

    prep_kernel<<<1, 256, 0, stream>>>(rots, trans, intrins, prots, ptrans, bias, camp, bpad);
    hipMemsetAsync(d_out, 0, (size_t)out_size * sizeof(float), stream);

    const int gemm_blocks = NCAM * O_TILES * HWG;                 // 1716
    gemm_wmma_kernel<<<gemm_blocks, 128, 0, stream>>>(img, W, bpad, xws);

    splat_kernel<<<NCAM * HW, 128, 0, stream>>>(xws, camp, out);
}
